// DifferentiableDiagAstar_23974507446677
// MI455X (gfx1250) — matrix operand order repaired and checked by reading, not compiled, not run on hardware
//
#include <hip/hip_runtime.h>
#include <hip/hip_bf16.h>

typedef __attribute__((ext_vector_type(16))) _Float16 v16h;
typedef __attribute__((ext_vector_type(8)))  float    v8f;

#define BATCH 32
#define SZ    64
#define NCELL 4096
#define TMAX  204                      // int(0.05 * 64 * 64)
#define R2    1.41421356237309515f     // float32(2**0.5) == 0x3FB504F3

// h(cell) = min(|dy|,|dx|)*sqrt2 + ||dy|-|dx||   (H_W == 1.0)
__device__ __forceinline__ float heur_val(int cell, float gy, float gx) {
  float dy = fabsf((float)(cell >> 6) - gy);
  float dx = fabsf((float)(cell & 63) - gx);
  return fminf(dy, dx) * R2 + fabsf(dy - dx);
}

// ---------------------------------------------------------------------------
// Zero the grid-sync area (barrier counter + 204 per-step solved bitmasks).
// Runs every call so graph replays are deterministic.
// ---------------------------------------------------------------------------
__global__ void __launch_bounds__(256) astar_init_ws(unsigned* gsync) {
  gsync[threadIdx.x] = 0u;   // words 0..255 : [0]=ctr, [1..204]=masks
}

// ---------------------------------------------------------------------------
// goal_loc = einsum('kij,bij->bk', loc, goal_maps) via v_wmma_f32_16x16x32_f16.
// A (16x32 f16): row 0 = y-coords, row 1 = x-coords, rows 2..15 = 0.
// B (32x16 f16): goal maps, batches 0-15 (acc0) and 16-31 (acc1).
// One-hot * integer-coordinate products are exact in f16/f32, so the result
// is the exact (gy, gx) per batch.  8 waves split the K=4096 reduction,
// partial C tiles summed through LDS.
// ---------------------------------------------------------------------------
__global__ void __launch_bounds__(256) astar_goal_wmma(const float* __restrict__ gm,
                                                       float* __restrict__ gyx) {
  const int tid  = threadIdx.x;
  const int lane = tid & 31;
  const int wv   = tid >> 5;             // 8 waves
  const int m    = lane & 15;            // A row / B column
  const int kbA  = (lane < 16) ? 0 : 8;  // 16-bit A 16x32 layout (ISA 7.12.2)
  const int kbB  = (lane < 16) ? 0 : 16; // unused; B uses A's K order (ka)

  v8f acc0 = {};
  v8f acc1 = {};
  for (int blk = 0; blk < 16; ++blk) {
    const int k0 = (wv * 16 + blk) << 5;     // this wave's K-block of 32
    v16h a, b0, b1;
#pragma unroll
    for (int e = 0; e < 16; ++e) {
      const int v = e >> 1, j = e & 1;
      // A: VGPR v<4 -> K = kbA + 2v + j ; v>=4 -> K = 16 + kbA + 2(v-4) + j
      const int ka    = (v < 4) ? (kbA + (v << 1) + j)
                                : (16 + kbA + ((v - 4) << 1) + j);
      const int cellA = k0 + ka;
      const float av  = (m == 0) ? (float)(cellA >> 6)
                      : (m == 1) ? (float)(cellA & 63) : 0.0f;
      a[e] = (_Float16)av;
      // B: same K as A (ka), column N = m
      const int cellB = k0 + ka;
      b0[e] = (_Float16)gm[m * NCELL + cellB];
      b1[e] = (_Float16)gm[(m + 16) * NCELL + cellB];
    }
    acc0 = __builtin_amdgcn_wmma_f32_16x16x32_f16(false, a, false, b0,
                                                  (short)0, acc0, false, false);
    acc1 = __builtin_amdgcn_wmma_f32_16x16x32_f16(false, a, false, b1,
                                                  (short)0, acc1, false, false);
  }

  __shared__ float sAcc[8][4][16];
  if (lane < 16) {                 // C layout: VGPR r, lanes 0-15 -> M=r, N=lane
    sAcc[wv][0][lane] = acc0[0];   // gy, batches 0-15
    sAcc[wv][1][lane] = acc0[1];   // gx, batches 0-15
    sAcc[wv][2][lane] = acc1[0];   // gy, batches 16-31
    sAcc[wv][3][lane] = acc1[1];   // gx, batches 16-31
  }
  __syncthreads();
  if (tid < 64) {
    const int c = tid >> 4, l = tid & 15;
    float s = 0.0f;
#pragma unroll
    for (int w = 0; w < 8; ++w) s += sAcc[w][c][l];
    const int batch = (c < 2) ? l : (l + 16);
    const int off   = ((c & 1) == 0) ? 0 : BATCH;   // gyx[0..31]=gy, [32..63]=gx
    gyx[off + batch] = s;
  }
}

// ---------------------------------------------------------------------------
// Core A* scan: one workgroup (8 wave32) per batch, all state LDS-resident.
// Per-step grid barrier across the 32 workgroups replicates the reference's
// global `done` flag (all batches select their goal in the same step).
// ---------------------------------------------------------------------------
__global__ void __launch_bounds__(256) astar_core_kernel(
    const float* __restrict__ start, const float* __restrict__ obst,
    const float* __restrict__ gyx, float* __restrict__ out, unsigned* gsync) {
  const int b   = blockIdx.x;
  const int tid = threadIdx.x;

  __shared__ float          sG[NCELL];      // 16 KB
  __shared__ float          sOM[NCELL];     // 16 KB
  __shared__ unsigned short sPar[NCELL];    //  8 KB
  __shared__ unsigned char  sOpen[NCELL];   //  4 KB
  __shared__ unsigned char  sHist[NCELL];   //  4 KB
  __shared__ unsigned char  sPath[NCELL];   //  4 KB
  __shared__ float sRedV[8];
  __shared__ int   sRedI[8];
  __shared__ int   sInd;
  __shared__ int   sFlag;

  const float gy = gyx[b], gx = gyx[BATCH + b];
  const int gidx = ((int)gy) * SZ + (int)gx;   // == argmax(goal) for one-hot goal

  for (int c = tid; c < NCELL; c += 256) {
    sG[c]    = 0.0f;
    sOM[c]   = obst[b * NCELL + c];
    sOpen[c] = (start[b * NCELL + c] != 0.0f) ? 1 : 0;
    sHist[c] = 0;
    sPar[c]  = (unsigned short)gidx;           // parents0 = goal_idx everywhere
  }
  __syncthreads();

  unsigned* ctr = gsync;
  unsigned* msk = gsync + 1;

  for (int t = 0; t < TMAX; ++t) {
    // ---- select: argmax over exp(-f/8)*open, first-index tie-break --------
    float bestv = -1.0f; int besti = 0;
    for (int j = 0; j < 16; ++j) {
      const int cell = tid + (j << 8);   // stride-256: bank-conflict-free
      const float f = 0.5f * sG[cell] + 0.501f * heur_val(cell, gy, gx);
      const float e = expf(-f * 0.125f) * (float)sOpen[cell];
      if (e > bestv) { bestv = e; besti = cell; }
    }
#pragma unroll
    for (int off = 16; off > 0; off >>= 1) {   // wave32 butterfly reduce
      const float ov = __shfl_xor(bestv, off, 32);
      const int   oi = __shfl_xor(besti, off, 32);
      if (ov > bestv || (ov == bestv && oi < besti)) { bestv = ov; besti = oi; }
    }
    if ((tid & 31) == 0) { sRedV[tid >> 5] = bestv; sRedI[tid >> 5] = besti; }
    __syncthreads();
    if (tid == 0) {
      float v = sRedV[0]; int i0 = sRedI[0];
      for (int w = 1; w < 8; ++w)
        if (sRedV[w] > v || (sRedV[w] == v && sRedI[w] < i0)) { v = sRedV[w]; i0 = sRedI[w]; }
      sInd = i0;
    }
    __syncthreads();
    const int ind = sInd;

    // ---- update: center (lane 8) + 8 neighbors (lanes 0..7) ---------------
    if (tid == 8) {
      sHist[ind] = 1;                          // hist2 = clip(hist + sel)
      if (ind != gidx) sOpen[ind] = 0;         // open2 = clip(open - unsolved*sel)
    } else if (tid < 8) {
      const int   d8y[8] = {-1,-1,-1, 0, 0, 1, 1, 1};
      const int   d8x[8] = {-1, 0, 1,-1, 1,-1, 0, 1};
      const float c8[8]  = {R2, 1.0f, R2, 1.0f, 1.0f, R2, 1.0f, R2};
      const int ny = (ind >> 6) + d8y[tid];
      const int nx = (ind & 63) + d8x[tid];
      if (ny >= 0 && ny < SZ && nx >= 0 && nx < SZ) {
        const int   n2  = (ny << 6) + nx;
        const float g2  = sG[ind] + c8[tid];   // g_sel + cost_filter
        const float op  = (float)sOpen[n2];
        const float hi  = (float)sHist[n2];
        const float cmp = (sG[n2] > g2) ? 1.0f : 0.0f;
        const float idxv = ((1.0f - op) * (1.0f - hi) + op * cmp) * sOM[n2];
        if (idxv != 0.0f) {
          sG[n2]    = g2 * idxv + sG[n2] * (1.0f - idxv);
          sOpen[n2] = 1;                       // clip(open + idx, 0, 1)
          sPar[n2]  = (unsigned short)ind;
        }
      }
    }
    __syncthreads();

    // ---- cross-batch `done` check (grid barrier over 32 workgroups) -------
    if (tid == 0) {
      if (ind == gidx) atomicOr(&msk[t], 1u << b);
      __threadfence();
      atomicAdd(ctr, 1u);
      const unsigned target = (unsigned)(BATCH * (t + 1));
      while (__hip_atomic_load(ctr, __ATOMIC_RELAXED, __HIP_MEMORY_SCOPE_AGENT) < target)
        __builtin_amdgcn_s_sleep(1);
      __threadfence();
      sFlag = (__hip_atomic_load(&msk[t], __ATOMIC_RELAXED,
                                 __HIP_MEMORY_SCOPE_AGENT) == 0xFFFFFFFFu);
    }
    __syncthreads();
    if (sFlag) break;   // done==true freezes all state in the reference
  }

  // ---- backtrack --------------------------------------------------------
  for (int c = tid; c < NCELL; c += 256) sPath[c] = (c == gidx) ? 1 : 0;
  __syncthreads();
  if (tid == 0) {
    int loc = sPar[gidx];                      // loc0 = parents[goal_idx]
    for (int t = 0; t < TMAX; ++t) { sPath[loc] = 1; loc = sPar[loc]; }
  }
  __syncthreads();

  // ---- outputs: [hist | path | g | h], each (B,1,64,64) f32 --------------
  float* oHist = out + 0 * (BATCH * NCELL) + b * NCELL;
  float* oPath = out + 1 * (BATCH * NCELL) + b * NCELL;
  float* oG    = out + 2 * (BATCH * NCELL) + b * NCELL;
  float* oH    = out + 3 * (BATCH * NCELL) + b * NCELL;
  for (int c = tid; c < NCELL; c += 256) {
    oHist[c] = (float)sHist[c];
    oPath[c] = (float)sPath[c];
    oG[c]    = sG[c];
    oH[c]    = heur_val(c, gy, gx);
  }
}

// ---------------------------------------------------------------------------
extern "C" void kernel_launch(void* const* d_in, const int* in_sizes, int n_in,
                              void* d_out, int out_size, void* d_ws, size_t ws_size,
                              hipStream_t stream) {
  (void)in_sizes; (void)n_in; (void)out_size; (void)ws_size;
  (void)d_in[0];                                   // cost_maps: unused by reference
  const float* start = (const float*)d_in[1];
  const float* goal  = (const float*)d_in[2];
  const float* obst  = (const float*)d_in[3];
  float*    out   = (float*)d_out;
  unsigned* gsync = (unsigned*)d_ws;               // [0]=ctr, [1..204]=step masks
  float*    gyx   = (float*)((char*)d_ws + 1024);  // 64 floats: gy[32] | gx[32]

  astar_init_ws  <<<1,     256, 0, stream>>>(gsync);
  astar_goal_wmma<<<1,     256, 0, stream>>>(goal, gyx);
  astar_core_kernel<<<BATCH, 256, 0, stream>>>(start, obst, gyx, out, gsync);
}
